// GNNConvNet_37538014167805
// MI455X (gfx1250) — compile-verified, matrix operand order checked by reading, not run on hardware
//
#include <hip/hip_runtime.h>
#include <hip/hip_bf16.h>

typedef __attribute__((ext_vector_type(16))) __bf16 v16bf;
typedef __attribute__((ext_vector_type(8)))  float  v8f;
typedef int v4i __attribute__((vector_size(16)));

#define CDIV(a,b) (((a)+(b)-1)/(b))

// Problem constants (match reference)
#define NN 50000
#define EE 800000
#define FF 64
#define DD 2
#define GG 50
#define CC 10

// ---------------------------------------------------------------------------
// 16-bit WMMA fragment K mapping (CDNA5 ISA 7.12.2, 16-bit A 16x32):
// lane 0-15: VGPR0..3 -> K 0..7 (pairs), VGPR4..7 -> K 16..23
// lane16-31: VGPR0..3 -> K 8..15,       VGPR4..7 -> K 24..31
__device__ __forceinline__ int kmap(int v, int hi) {
  int base = (v < 4) ? (2 * v) : (16 + 2 * (v - 4));
  return base + hi * 8;
}

__device__ __forceinline__ v8f wmma_bf16(v16bf a, v16bf b, v8f c) {
  return __builtin_amdgcn_wmma_f32_16x16x32_bf16(false, a, false, b, (short)0, c,
                                                 false, false);
}

// ---------------------------------------------------------------------------
// Async global->LDS staging (CDNA5 GLOBAL_LOAD_ASYNC_TO_LDS_B128, ASYNCcnt),
// with a synchronous fallback so compilation never depends on the builtin.
#if __has_builtin(__builtin_amdgcn_global_load_async_to_lds_b128)
#define HAVE_ASYNC_LDS 1
#endif

__device__ __forceinline__ void stage_g2l(const void* g, void* l, int bytes,
                                          int tid, int nthr) {
#ifdef HAVE_ASYNC_LDS
  __attribute__((address_space(1))) v4i* gp =
      (__attribute__((address_space(1))) v4i*)(g);
  __attribute__((address_space(3))) v4i* lp =
      (__attribute__((address_space(3))) v4i*)(l);
  for (int o = tid; o * 16 < bytes; o += nthr) {
    __builtin_amdgcn_global_load_async_to_lds_b128(gp + o, lp + o, 0, 0);
  }
#else
  const int4* gs = (const int4*)g;
  int4* ls = (int4*)l;
  for (int i = tid; i * 16 < bytes; i += nthr) ls[i] = gs[i];
#endif
}

__device__ __forceinline__ void stage_wait() {
#if __has_builtin(__builtin_amdgcn_s_wait_asynccnt)
  __builtin_amdgcn_s_wait_asynccnt(0);
#elif defined(HAVE_ASYNC_LDS)
  asm volatile("s_wait_asynccnt 0" ::: "memory");
#endif
}

// ---------------------------------------------------------------------------
// Pack fp32 weight [K, Ncol] into bf16 B-fragments.
// Fragment (chunk c, colTile t): 32 lanes x 16 bf16.
__global__ void k_pack_bf16(const float* __restrict__ W, __bf16* __restrict__ out,
                            int K, int Ncol, int nChunks, int nColTiles) {
  int tid = blockIdx.x * blockDim.x + threadIdx.x;
  int total = nChunks * nColTiles * 32;
  if (tid >= total) return;
  int lane = tid & 31;
  int ft = tid >> 5;
  int t = ft % nColTiles;
  int c = ft / nColTiles;
  int n = t * 16 + (lane & 15);
  int hi = lane >> 4;
  __bf16* dst = out + (size_t)tid * 16;
#pragma unroll
  for (int v = 0; v < 8; ++v) {
    int k0 = c * 32 + kmap(v, hi);
    float w0 = (k0 < K && n < Ncol) ? W[(size_t)k0 * Ncol + n] : 0.f;
    float w1 = (k0 + 1 < K && n < Ncol) ? W[(size_t)(k0 + 1) * Ncol + n] : 0.f;
    dst[2 * v] = (__bf16)w0;
    dst[2 * v + 1] = (__bf16)w1;
  }
}

// ---------------------------------------------------------------------------
// CGConv edge kernel: 128 threads = 4 waves, 16 edges/wave (E % 64 == 0).
// z = [x[dst], x[src], ea] (130 -> pad 160), bf16 in LDS.
// msg = sigmoid(z@Wf+bf) * softplus(z@Ws+bs); atomicAdd into out[dst].
// Both packed weight matrices (20 KB each) async-staged to LDS per block.
#define CG_KP 160
#define CG_WV 4
__global__ void k_cgconv_edge(const float* __restrict__ x,
                              const int* __restrict__ src, const int* __restrict__ dst,
                              const float* __restrict__ ea,
                              const __bf16* __restrict__ Wfp, const __bf16* __restrict__ Wsp,
                              const float* __restrict__ bf, const float* __restrict__ bs,
                              float* __restrict__ out) {
  __shared__ __align__(16) __bf16 zt[CG_WV][16 * CG_KP];   // 20480 B
  __shared__ __align__(16) __bf16 wfl[5 * 4 * 512];        // 20480 B
  __shared__ __align__(16) __bf16 wsl[5 * 4 * 512];        // 20480 B
  __shared__ int dse[CG_WV][16];
  int tid = threadIdx.x;
  int lane = tid & 31;
  int w = tid >> 5;
  int base = (blockIdx.x * CG_WV + w) * 16;

  // async-stage packed weights (shared by all 4 waves)
  stage_g2l(Wfp, wfl, 5 * 4 * 512 * 2, tid, 128);
  stage_g2l(Wsp, wsl, 5 * 4 * 512 * 2, tid, 128);

  if (lane < 16) dse[w][lane] = dst[base + lane];
  // gather 16x160 bf16 tile (coalesced along k)
  for (int i = 0; i < 80; ++i) {
    int idx = i * 32 + lane;
    int e = idx / CG_KP;
    int k = idx - e * CG_KP;
    int eg = base + e;
    float val;
    if (k < 64)        val = x[(size_t)dst[eg] * FF + k];
    else if (k < 128)  val = x[(size_t)src[eg] * FF + (k - 64)];
    else if (k < 130)  val = ea[(size_t)eg * DD + (k - 128)];
    else               val = 0.f;
    zt[w][e * CG_KP + k] = (__bf16)val;
  }
  stage_wait();
  __syncthreads();

  int m = lane & 15, hi = lane >> 4;
  // A fragments (5 K-chunks), reused across all 8 output tiles
  v16bf afr[5];
  {
    const __bf16* row = &zt[w][m * CG_KP];
#pragma unroll
    for (int c = 0; c < 5; ++c)
#pragma unroll
      for (int v = 0; v < 8; ++v) {
        int k0 = c * 32 + kmap(v, hi);
        afr[c][2 * v] = row[k0];
        afr[c][2 * v + 1] = row[k0 + 1];
      }
  }
  int de[8];
#pragma unroll
  for (int r = 0; r < 8; ++r) de[r] = dse[w][r + hi * 8];

#pragma unroll
  for (int t = 0; t < 4; ++t) {
    v8f accf = {}, accs = {};
#pragma unroll
    for (int c = 0; c < 5; ++c) {
      v16bf bfr = *(const v16bf*)(wfl + ((c * 4 + t) * 32 + lane) * 16);
      accf = wmma_bf16(afr[c], bfr, accf);
      v16bf bsr = *(const v16bf*)(wsl + ((c * 4 + t) * 32 + lane) * 16);
      accs = wmma_bf16(afr[c], bsr, accs);
    }
    int n = t * 16 + m;
    float bfv = bf[n], bsv = bs[n];
#pragma unroll
    for (int r = 0; r < 8; ++r) {
      float fg = accf[r] + bfv;
      float sg = accs[r] + bsv;
      float sig = __frcp_rn(1.f + __expf(-fg));
      // stable fast softplus: max(x,0) + log(1 + exp(-|x|))
      float sp = fmaxf(sg, 0.f) + __logf(1.f + __expf(-fabsf(sg)));
      atomicAdd(&out[(size_t)de[r] * FF + n], sig * sp);
    }
  }
}

// ---------------------------------------------------------------------------
// Generic wave-tile GEMM: out[:, colOff + n] = act(A[M,K] @ Wp + bias)
// Packed weight fragments async-staged into LDS (<= 12 fragments).
__global__ void k_gemm_bf16(const float* __restrict__ A, int M, int K, int lda,
                            const __bf16* __restrict__ Bp, int nChunks, int nColTiles,
                            const float* __restrict__ bias, float* __restrict__ out,
                            int ldo, int colOff, int Ncol, int act) {
  __shared__ __align__(16) __bf16 wlds[12 * 512];  // up to 12 KB of fragments
  int tid = threadIdx.x;
  int lane = tid & 31;
  int nFrag = nChunks * nColTiles;
  stage_g2l(Bp, wlds, nFrag * 512 * 2, tid, blockDim.x);
  stage_wait();
  __syncthreads();

  int wave = blockIdx.x * (blockDim.x >> 5) + (tid >> 5);
  int rowTiles = (M + 15) >> 4;
  if (wave >= rowTiles * nColTiles) return;  // wave-uniform, after barrier
  int rt = wave / nColTiles, ct = wave % nColTiles;
  int m = lane & 15, hi = lane >> 4;
  int row = rt * 16 + m;
  if (row >= M) row = M - 1;  // clamp (stores are guarded)
  const float* arow = A + (size_t)row * lda;

  v8f acc = {};
  for (int c = 0; c < nChunks; ++c) {
    v16bf a;
#pragma unroll
    for (int v = 0; v < 8; ++v) {
      int k0 = c * 32 + kmap(v, hi);
      float x0 = (k0 < K) ? arow[k0] : 0.f;
      float x1 = (k0 + 1 < K) ? arow[k0 + 1] : 0.f;
      a[2 * v] = (__bf16)x0;
      a[2 * v + 1] = (__bf16)x1;
    }
    v16bf b = *(const v16bf*)(wlds + ((c * nColTiles + ct) * 32 + lane) * 16);
    acc = wmma_bf16(a, b, acc);
  }
  int n = ct * 16 + m;
  float bv = (bias != nullptr && n < Ncol) ? bias[n] : 0.f;
#pragma unroll
  for (int r = 0; r < 8; ++r) {
    int orow = rt * 16 + r + hi * 8;
    if (orow < M && n < Ncol) {
      float v = acc[r] + bv;
      if (act == 1) v = fmaxf(v, 0.f);
      out[(size_t)orow * ldo + colOff + n] = v;
    }
  }
}

// ---------------------------------------------------------------------------
__global__ void k_fill(float* p, float v, long n) {
  long i = (long)blockIdx.x * blockDim.x + threadIdx.x;
  if (i < n) p[i] = v;
}

__global__ void k_bias_relu(float* h, const float* bias, int Nn, int Fc) {
  long i = (long)blockIdx.x * blockDim.x + threadIdx.x;
  if (i >= (long)Nn * Fc) return;
  int f = (int)(i % Fc);
  float v = h[i] + (bias ? bias[f] : 0.f);
  h[i] = fmaxf(v, 0.f);
}

__device__ __forceinline__ void atomicMaxF(float* addr, float val) {
  int* ai = (int*)addr;
  int old = __float_as_int(*(volatile float*)addr);
  while (__int_as_float(old) < val) {
    int prev = atomicCAS(ai, old, __float_as_int(val));
    if (prev == old) break;
    old = prev;
  }
}

// ---------------------------------------------------------------------------
// GAT attention
__global__ void k_gat_coef(const float* __restrict__ h, const float* __restrict__ asrc,
                           const float* __restrict__ adst, float* a_s, float* a_d,
                           int Nn, int heads, int oc) {
  int tid = blockIdx.x * blockDim.x + threadIdx.x;
  if (tid >= Nn * heads) return;
  int n = tid / heads, hd = tid % heads;
  const float* hp = h + (size_t)n * heads * oc + hd * oc;
  float ss = 0.f, sd = 0.f;
  for (int c = 0; c < oc; ++c) {
    float v = hp[c];
    ss += v * asrc[hd * oc + c];
    sd += v * adst[hd * oc + c];
  }
  a_s[tid] = ss;
  a_d[tid] = sd;
}

__device__ __forceinline__ void gat_edge_sd(int e, const int* src, const int* dst,
                                            int E, int* s, int* d) {
  if (e < E) { *s = src[e]; *d = dst[e]; }
  else { *s = e - E; *d = e - E; }  // self loops
}

__global__ void k_gat_edge_max(const int* src, const int* dst, const float* a_s,
                               const float* a_d, float* emax, int E, int Nn, int heads) {
  int tid = blockIdx.x * blockDim.x + threadIdx.x;
  if (tid >= (E + Nn) * heads) return;
  int e = tid / heads, hd = tid % heads;
  int s, d;
  gat_edge_sd(e, src, dst, E, &s, &d);
  float ev = a_s[s * heads + hd] + a_d[d * heads + hd];
  ev = (ev > 0.f) ? ev : 0.2f * ev;
  atomicMaxF(&emax[d * heads + hd], ev);
}

__global__ void k_gat_edge_exp(const int* src, const int* dst, const float* a_s,
                               const float* a_d, const float* emax, float* exbuf,
                               float* denom, int E, int Nn, int heads) {
  int tid = blockIdx.x * blockDim.x + threadIdx.x;
  if (tid >= (E + Nn) * heads) return;
  int e = tid / heads, hd = tid % heads;
  int s, d;
  gat_edge_sd(e, src, dst, E, &s, &d);
  float ev = a_s[s * heads + hd] + a_d[d * heads + hd];
  ev = (ev > 0.f) ? ev : 0.2f * ev;
  float ex = __expf(ev - emax[d * heads + hd]);
  exbuf[tid] = ex;
  atomicAdd(&denom[d * heads + hd], ex);
}

// normalize: exbuf[e,h] <- exbuf[e,h] / (denom[d,h] + 1e-16)
__global__ void k_gat_alpha(const int* src, const int* dst, const float* __restrict__ denom,
                            float* exbuf, int E, int Nn, int heads) {
  int tid = blockIdx.x * blockDim.x + threadIdx.x;
  if (tid >= (E + Nn) * heads) return;
  int e = tid / heads, hd = tid % heads;
  int s, d;
  gat_edge_sd(e, src, dst, E, &s, &d);
  exbuf[tid] = exbuf[tid] / (denom[d * heads + hd] + 1e-16f);
}

__global__ void k_gat_edge_out(const int* src, const int* dst, const float* __restrict__ h,
                               const float* __restrict__ alpha, float* out,
                               int E, int Nn, int heads, int oc) {
  long tid = (long)blockIdx.x * blockDim.x + threadIdx.x;
  int hc = heads * oc;
  if (tid >= (long)(E + Nn) * hc) return;
  int e = (int)(tid / hc);
  int r = (int)(tid % hc);
  int hd = r / oc;
  int s, d;
  gat_edge_sd(e, src, dst, E, &s, &d);
  atomicAdd(&out[(size_t)d * hc + r],
            alpha[(size_t)e * heads + hd] * h[(size_t)s * hc + r]);
}

// ---------------------------------------------------------------------------
// SGConv (K=1, gcn_norm)
__global__ void k_sg_deg(const int* dst, const float* ea, float* deg, int E) {
  int tid = blockIdx.x * blockDim.x + threadIdx.x;
  if (tid >= E * 2) return;
  int e = tid >> 1, c = tid & 1;
  atomicAdd(&deg[(size_t)dst[e] * 2 + c], fabsf(ea[(size_t)e * 2 + c]));
}

__global__ void k_sg_dinv(const float* deg, float* dinv, int n) {
  int i = blockIdx.x * blockDim.x + threadIdx.x;
  if (i >= n) return;
  float d = deg[i];
  dinv[i] = (d > 0.f) ? rsqrtf(fmaxf(d, 1e-12f)) : 0.f;
}

__global__ void k_sg_agg(const float* __restrict__ xin, const int* src, const int* dst,
                         const float* ea, const float* __restrict__ dinv,
                         float* agg, int E, int Nn, int K, int ch) {
  long tid = (long)blockIdx.x * blockDim.x + threadIdx.x;
  if (tid >= (long)(E + Nn) * K) return;
  int e = (int)(tid / K);
  int k = (int)(tid % K);
  if (e < E) {
    int s = src[e], d = dst[e];
    float w = fabsf(ea[(size_t)e * 2 + ch]);
    float nrm = dinv[s * 2 + ch] * w * dinv[d * 2 + ch];
    atomicAdd(&agg[(size_t)d * K + k], nrm * xin[(size_t)s * K + k]);
  } else {
    int n = e - E;
    float di = dinv[n * 2 + ch];
    atomicAdd(&agg[(size_t)n * K + k], di * di * xin[(size_t)n * K + k]);
  }
}

// ---------------------------------------------------------------------------
__global__ void k_pool(const float* __restrict__ h, const int* __restrict__ batch,
                       float* pooled, float* cnt, int Nn, int K) {
  long tid = (long)blockIdx.x * blockDim.x + threadIdx.x;
  if (tid >= (long)Nn * K) return;
  int n = (int)(tid / K);
  int k = (int)(tid % K);
  int g = batch[n];
  atomicAdd(&pooled[(size_t)g * K + k], h[(size_t)n * K + k]);
  if (k == 0) atomicAdd(&cnt[g], 1.f);
}

__global__ void k_pool_div(float* pooled, const float* cnt, int G, int K) {
  int tid = blockIdx.x * blockDim.x + threadIdx.x;
  if (tid >= G * K) return;
  pooled[tid] /= fmaxf(cnt[tid / K], 1.f);
}

// ---------------------------------------------------------------------------
extern "C" void kernel_launch(void* const* d_in, const int* in_sizes, int n_in,
                              void* d_out, int out_size, void* d_ws, size_t ws_size,
                              hipStream_t stream) {
  const float* x      = (const float*)d_in[0];
  const int*   eidx   = (const int*)d_in[1];
  const float* eattr  = (const float*)d_in[2];
  const int*   batch  = (const int*)d_in[3];
  const float* cg1_fW = (const float*)d_in[4];  const float* cg1_fb = (const float*)d_in[5];
  const float* cg1_sW = (const float*)d_in[6];  const float* cg1_sb = (const float*)d_in[7];
  const float* cg2_fW = (const float*)d_in[8];  const float* cg2_fb = (const float*)d_in[9];
  const float* cg2_sW = (const float*)d_in[10]; const float* cg2_sb = (const float*)d_in[11];
  const float* gat1_W = (const float*)d_in[12]; const float* gat1_as = (const float*)d_in[13];
  const float* gat1_ad = (const float*)d_in[14]; const float* gat1_b = (const float*)d_in[15];
  const float* gat2_W = (const float*)d_in[16]; const float* gat2_as = (const float*)d_in[17];
  const float* gat2_ad = (const float*)d_in[18]; const float* gat2_b = (const float*)d_in[19];
  const float* sg1a_W = (const float*)d_in[20]; const float* sg1a_b = (const float*)d_in[21];
  const float* sg1b_W = (const float*)d_in[22]; const float* sg1b_b = (const float*)d_in[23];
  const float* sg2a_W = (const float*)d_in[24]; const float* sg2a_b = (const float*)d_in[25];
  const float* sg2b_W = (const float*)d_in[26]; const float* sg2b_b = (const float*)d_in[27];
  const float* lin1_W = (const float*)d_in[28]; const float* lin1_b = (const float*)d_in[29];
  const float* lin2_W = (const float*)d_in[30]; const float* lin2_b = (const float*)d_in[31];

  const int* src = eidx;
  const int* dst = eidx + EE;
  float* out = (float*)d_out;

  // ---- workspace carve-out ------------------------------------------------
  size_t off = 0;
  auto alloc = [&](size_t bytes) -> void* {
    void* p = (char*)d_ws + off;
    off += (bytes + 255) & ~(size_t)255;
    return p;
  };
  float* A   = (float*)alloc((size_t)NN * 96 * 4);
  float* B   = (float*)alloc((size_t)NN * 96 * 4);
  float* Cb  = (float*)alloc((size_t)NN * 96 * 4);
  float* Db  = (float*)alloc((size_t)NN * 64 * 4);
  float* a_s = (float*)alloc((size_t)NN * 3 * 4);
  float* a_d = (float*)alloc((size_t)NN * 3 * 4);
  float* emax  = (float*)alloc((size_t)NN * 3 * 4);
  float* denom = (float*)alloc((size_t)NN * 3 * 4);
  float* exbuf = (float*)alloc((size_t)(EE + NN) * 3 * 4);
  float* deg   = (float*)alloc((size_t)NN * 2 * 4);
  float* dinv  = (float*)alloc((size_t)NN * 2 * 4);
  float* pooled = (float*)alloc((size_t)GG * 64 * 4);
  float* cnt    = (float*)alloc((size_t)GG * 4);
  float* hh     = (float*)alloc((size_t)GG * 16 * 4);
  __bf16* p_cg1f = (__bf16*)alloc(5 * 4 * 512 * 2);
  __bf16* p_cg1s = (__bf16*)alloc(5 * 4 * 512 * 2);
  __bf16* p_cg2f = (__bf16*)alloc(5 * 4 * 512 * 2);
  __bf16* p_cg2s = (__bf16*)alloc(5 * 4 * 512 * 2);
  __bf16* p_g1W  = (__bf16*)alloc(2 * 6 * 512 * 2);
  __bf16* p_g2W  = (__bf16*)alloc(3 * 3 * 512 * 2);
  __bf16* p_s1a  = (__bf16*)alloc(2 * 2 * 512 * 2);
  __bf16* p_s1b  = (__bf16*)alloc(2 * 2 * 512 * 2);
  __bf16* p_s2a  = (__bf16*)alloc(2 * 2 * 512 * 2);
  __bf16* p_s2b  = (__bf16*)alloc(2 * 2 * 512 * 2);
  __bf16* p_l1   = (__bf16*)alloc(2 * 1 * 512 * 2);
  __bf16* p_l2   = (__bf16*)alloc(1 * 1 * 512 * 2);

  auto pack = [&](const float* W, __bf16* p, int K, int Nc, int ch, int ct) {
    int tot = ch * ct * 32;
    k_pack_bf16<<<CDIV(tot, 256), 256, 0, stream>>>(W, p, K, Nc, ch, ct);
  };
  pack(cg1_fW, p_cg1f, 130, 64, 5, 4);  pack(cg1_sW, p_cg1s, 130, 64, 5, 4);
  pack(cg2_fW, p_cg2f, 130, 64, 5, 4);  pack(cg2_sW, p_cg2s, 130, 64, 5, 4);
  pack(gat1_W, p_g1W, 64, 96, 2, 6);    pack(gat2_W, p_g2W, 96, 48, 3, 3);
  pack(sg1a_W, p_s1a, 48, 32, 2, 2);    pack(sg1b_W, p_s1b, 48, 32, 2, 2);
  pack(sg2a_W, p_s2a, 64, 32, 2, 2);    pack(sg2b_W, p_s2b, 64, 32, 2, 2);
  pack(lin1_W, p_l1, 64, 16, 2, 1);     pack(lin2_W, p_l2, 16, 10, 1, 1);

  auto fill = [&](float* p, float v, long n) {
    k_fill<<<CDIV(n, 256), 256, 0, stream>>>(p, v, n);
  };
  auto gemm = [&](const float* Ain, int M, int K, int lda, const __bf16* Bp,
                  int ch, int ct, const float* bias, float* o, int ldo, int co,
                  int Nc, int act) {
    int waves = CDIV(M, 16) * ct;
    k_gemm_bf16<<<CDIV(waves, 8), 256, 0, stream>>>(Ain, M, K, lda, Bp, ch, ct,
                                                    bias, o, ldo, co, Nc, act);
  };

  const int cgBlocks = EE / (16 * CG_WV);  // 64 edges per 128-thread block
  // ---- CGConv 1: A = relu(x + scatter(msg)) ------------------------------
  (void)hipMemcpyAsync(A, x, (size_t)NN * FF * 4, hipMemcpyDeviceToDevice, stream);
  k_cgconv_edge<<<cgBlocks, 128, 0, stream>>>(x, src, dst, eattr, p_cg1f, p_cg1s,
                                              cg1_fb, cg1_sb, A);
  k_bias_relu<<<CDIV((long)NN * 64, 256), 256, 0, stream>>>(A, nullptr, NN, 64);
  // ---- CGConv 2: B -------------------------------------------------------
  (void)hipMemcpyAsync(B, A, (size_t)NN * FF * 4, hipMemcpyDeviceToDevice, stream);
  k_cgconv_edge<<<cgBlocks, 128, 0, stream>>>(A, src, dst, eattr, p_cg2f, p_cg2s,
                                              cg2_fb, cg2_sb, B);
  k_bias_relu<<<CDIV((long)NN * 64, 256), 256, 0, stream>>>(B, nullptr, NN, 64);

  // ---- GAT 1: h(Cb) = B @ W;  out -> A (96) ------------------------------
  gemm(B, NN, 64, 64, p_g1W, 2, 6, nullptr, Cb, 96, 0, 96, 0);
  k_gat_coef<<<CDIV(NN * 3, 256), 256, 0, stream>>>(Cb, gat1_as, gat1_ad, a_s, a_d, NN, 3, 32);
  fill(emax, -1e30f, (long)NN * 3);
  fill(denom, 0.f, (long)NN * 3);
  fill(A, 0.f, (long)NN * 96);
  {
    long t1 = (long)(EE + NN) * 3;
    k_gat_edge_max<<<CDIV(t1, 256), 256, 0, stream>>>(src, dst, a_s, a_d, emax, EE, NN, 3);
    k_gat_edge_exp<<<CDIV(t1, 256), 256, 0, stream>>>(src, dst, a_s, a_d, emax, exbuf, denom, EE, NN, 3);
    k_gat_alpha<<<CDIV(t1, 256), 256, 0, stream>>>(src, dst, denom, exbuf, EE, NN, 3);
    long t2 = (long)(EE + NN) * 96;
    k_gat_edge_out<<<CDIV(t2, 256), 256, 0, stream>>>(src, dst, Cb, exbuf, A, EE, NN, 3, 32);
  }
  k_bias_relu<<<CDIV((long)NN * 96, 256), 256, 0, stream>>>(A, gat1_b, NN, 96);

  // ---- GAT 2: h(B,48) = A @ W2; out -> Cb (48) ---------------------------
  gemm(A, NN, 96, 96, p_g2W, 3, 3, nullptr, B, 48, 0, 48, 0);
  k_gat_coef<<<CDIV(NN * 3, 256), 256, 0, stream>>>(B, gat2_as, gat2_ad, a_s, a_d, NN, 3, 16);
  fill(emax, -1e30f, (long)NN * 3);
  fill(denom, 0.f, (long)NN * 3);
  fill(Cb, 0.f, (long)NN * 48);
  {
    long t1 = (long)(EE + NN) * 3;
    k_gat_edge_max<<<CDIV(t1, 256), 256, 0, stream>>>(src, dst, a_s, a_d, emax, EE, NN, 3);
    k_gat_edge_exp<<<CDIV(t1, 256), 256, 0, stream>>>(src, dst, a_s, a_d, emax, exbuf, denom, EE, NN, 3);
    k_gat_alpha<<<CDIV(t1, 256), 256, 0, stream>>>(src, dst, denom, exbuf, EE, NN, 3);
    long t2 = (long)(EE + NN) * 48;
    k_gat_edge_out<<<CDIV(t2, 256), 256, 0, stream>>>(src, dst, B, exbuf, Cb, EE, NN, 3, 16);
  }
  k_bias_relu<<<CDIV((long)NN * 48, 256), 256, 0, stream>>>(Cb, gat2_b, NN, 48);

  // ---- SGConv shared gcn-norm --------------------------------------------
  fill(deg, 1.0f, (long)NN * 2);  // self-loop weight
  k_sg_deg<<<CDIV(EE * 2, 256), 256, 0, stream>>>(dst, eattr, deg, EE);
  k_sg_dinv<<<CDIV(NN * 2, 256), 256, 0, stream>>>(deg, dinv, NN * 2);

  // ---- SG layer 1: input Cb(48) -> Cb(64) --------------------------------
  fill(A, 0.f, (long)NN * 48);
  fill(B, 0.f, (long)NN * 48);
  {
    long t = (long)(EE + NN) * 48;
    k_sg_agg<<<CDIV(t, 256), 256, 0, stream>>>(Cb, src, dst, eattr, dinv, A, EE, NN, 48, 0);
    k_sg_agg<<<CDIV(t, 256), 256, 0, stream>>>(Cb, src, dst, eattr, dinv, B, EE, NN, 48, 1);
  }
  gemm(A, NN, 48, 48, p_s1a, 2, 2, sg1a_b, Cb, 64, 0, 32, 1);
  gemm(B, NN, 48, 48, p_s1b, 2, 2, sg1b_b, Cb, 64, 32, 32, 1);

  // ---- SG layer 2: input Cb(64) -> Db(64) --------------------------------
  fill(A, 0.f, (long)NN * 64);
  fill(B, 0.f, (long)NN * 64);
  {
    long t = (long)(EE + NN) * 64;
    k_sg_agg<<<CDIV(t, 256), 256, 0, stream>>>(Cb, src, dst, eattr, dinv, A, EE, NN, 64, 0);
    k_sg_agg<<<CDIV(t, 256), 256, 0, stream>>>(Cb, src, dst, eattr, dinv, B, EE, NN, 64, 1);
  }
  gemm(A, NN, 64, 64, p_s2a, 2, 2, sg2a_b, Db, 64, 0, 32, 1);
  gemm(B, NN, 64, 64, p_s2b, 2, 2, sg2b_b, Db, 64, 32, 32, 1);

  // ---- mean pool + head ---------------------------------------------------
  fill(pooled, 0.f, (long)GG * 64);
  fill(cnt, 0.f, GG);
  k_pool<<<CDIV((long)NN * 64, 256), 256, 0, stream>>>(Db, batch, pooled, cnt, NN, 64);
  k_pool_div<<<CDIV(GG * 64, 256), 256, 0, stream>>>(pooled, cnt, GG, 64);
  gemm(pooled, GG, 64, 64, p_l1, 2, 1, lin1_b, hh, 16, 0, 16, 1);
  gemm(hh, GG, 16, 16, p_l2, 1, 1, lin2_b, out, CC, 0, CC, 0);
}
